// SPOP_25056839206032
// MI455X (gfx1250) — compile-verified, matrix operand order checked by reading, not run on hardware
//
#include <hip/hip_runtime.h>
#include <hip/hip_bf16.h>
#include <stdint.h>

// Problem constants (from reference)
#define Nn 512
#define Ss 64
#define Cc 1024
#define Bb 10
#define NEGV 1e9f

typedef __attribute__((ext_vector_type(16))) _Float16 v16h;
typedef __attribute__((ext_vector_type(8)))  float    v8f;
typedef __attribute__((ext_vector_type(4)))  unsigned uint32x4;
typedef __attribute__((ext_vector_type(4)))  int      int32x4;
typedef __attribute__((ext_vector_type(8)))  int      int32x8;

// ---------------------------------------------------------------------------
// Kernel 1: per-row bincount with "drop last non-pad" semantics.
// One block per n. counts[n, c] written to workspace as float.
// ---------------------------------------------------------------------------
__global__ void counts_kernel(const int* __restrict__ item_ids,
                              float* __restrict__ counts) {
    __shared__ int hist[Cc];
    __shared__ int last_pos;
    const int n = blockIdx.x;
    const int t = threadIdx.x;               // blockDim = 128
    if (t == 0) last_pos = -1;
    for (int c = t; c < Cc; c += 128) hist[c] = 0;
    __syncthreads();

    int id = 0;
    if (t < Ss) {
        id = item_ids[n * Ss + t];
        if (id != 0) atomicMax(&last_pos, t);   // last non-pad position
    }
    __syncthreads();
    if (t < Ss && id != 0 && t != last_pos)     // all-pad row: no adds at all
        atomicAdd(&hist[id], 1);
    __syncthreads();

    for (int c = t; c < Cc; c += 128)
        counts[(size_t)n * Cc + c] = (float)hist[c];
}

// ---------------------------------------------------------------------------
// Kernel 2: masked log_softmax, one 512-thread block (16 waves) per n.
//   counts row staged global->LDS via TDM (async tensor path).
//   Ban ids expanded once into an LDS bitmask (set-semantics for free).
//   Phase A: per-s scalar = max + log(sum exp(masked)), branchless cndmask.
//   Phase B: 16x16 output tiles via v_wmma_f32_16x16x32_f16 broadcast,
//            branchless ban patch in accumulator layout, stream to global.
// ---------------------------------------------------------------------------
__global__ void softmax_kernel(const float* __restrict__ counts,
                               const int* __restrict__ ban_ids,
                               float* __restrict__ out) {
    __shared__ __align__(16) float cnt[Cc];   // 4 KB counts row
    __shared__ unsigned bmask[Ss * 32];       // 8 KB: 1024-bit ban mask per s
    __shared__ float scal[Ss];                // per-s (max + lse)

    const int n    = blockIdx.x;
    const int tid  = threadIdx.x;
    const int wave = tid >> 5;
    const int lane = tid & 31;

    // --- TDM: counts[n, 0:1024] (4KB) global -> LDS, issued by wave 0 only ---
    if (wave == 0) {
        unsigned lds_addr = (unsigned)(uintptr_t)(void*)&cnt[0];  // LDS aperture: low 32b
        unsigned long long ga =
            (unsigned long long)(uintptr_t)(counts + (size_t)n * Cc);
        uint32x4 g0;
        g0[0] = 1u;                                   // count=1 (valid descriptor)
        g0[1] = lds_addr;                             // lds_addr [63:32]
        g0[2] = (unsigned)ga;                         // global_addr [95:64]
        g0[3] = (unsigned)((ga >> 32) & 0x01ffffffull) | (2u << 30); // addr[56:32], type=2
        int32x8 g1;
        g1[0] = (int)(2u << 16);      // data_size=2 (4 bytes)
        g1[1] = (int)(1024u << 16);   // tensor_dim0 = 1024 (low16 at bits 63:48)
        g1[2] = (int)(1u << 16);      // tensor_dim0 hi=0 ; tensor_dim1 = 1
        g1[3] = (int)(1024u << 16);   // tensor_dim1 hi=0 ; tile_dim0 = 1024
        g1[4] = 1;                    // tile_dim1 = 1, tile_dim2 = 0
        g1[5] = 1024;                 // tensor_dim0_stride = 1024
        g1[6] = (int)(1024u << 16);   // stride0 hi=0 ; tensor_dim1_stride = 1024
        g1[7] = 0;
        int32x4 gz; gz[0] = gz[1] = gz[2] = gz[3] = 0;
#if __has_include(<hip/amd_detail/amd_gfx1250_TDM.h>)
        int32x8 gx; gx[0]=gx[1]=gx[2]=gx[3]=gx[4]=gx[5]=gx[6]=gx[7]=0;
        __builtin_amdgcn_tensor_load_to_lds(g0, g1, gz, gz, gx, 0);
#else
        __builtin_amdgcn_tensor_load_to_lds(g0, g1, gz, gz, 0);
#endif
    }

    // --- zero the ban bitmask (2048 words), then scatter bans into it ---
    for (int i = tid; i < Ss * 32; i += 512) bmask[i] = 0u;
    if (wave == 0) __builtin_amdgcn_s_wait_tensorcnt(0);  // cnt ready for phases
    __syncthreads();

    for (int i = tid; i < Ss * Bb; i += 512) {
        const int s = i / Bb;
        const int c = ban_ids[(size_t)n * Ss * Bb + i];
        atomicOr(&bmask[s * 32 + (c >> 5)], 1u << (c & 31));
    }
    __syncthreads();

    // --- Phase A: per-s max + log-sum-exp over masked counts row ---
    const float4* cnt4 = reinterpret_cast<const float4*>(cnt);
    for (int s = wave; s < Ss; s += 16) {                // 4 rows per wave
        const unsigned* mrow = &bmask[s * 32];
        const int bitbase = (lane & 7) * 4;              // c & 31 for j=0
        float4 vv[8];
        float m = -3.0e38f;
        #pragma unroll
        for (int k = 0; k < 8; ++k) {
            // c = k*128 + lane*4 + j  (j = 0..3), fully coalesced b128 loads
            float4 v4 = cnt4[k * 32 + lane];
            const unsigned w = mrow[k * 4 + (lane >> 3)];
            v4.x = ((w >> (bitbase + 0)) & 1u) ? v4.x - NEGV : v4.x;
            v4.y = ((w >> (bitbase + 1)) & 1u) ? v4.y - NEGV : v4.y;
            v4.z = ((w >> (bitbase + 2)) & 1u) ? v4.z - NEGV : v4.z;
            v4.w = ((w >> (bitbase + 3)) & 1u) ? v4.w - NEGV : v4.w;
            vv[k] = v4;
            m = fmaxf(m, fmaxf(fmaxf(v4.x, v4.y), fmaxf(v4.z, v4.w)));
        }
        #pragma unroll
        for (int off = 16; off > 0; off >>= 1)
            m = fmaxf(m, __shfl_xor(m, off, 32));
        float sum = 0.0f;
        #pragma unroll
        for (int k = 0; k < 8; ++k) {
            sum += __expf(vv[k].x - m);                  // banned -> exp==0 (fp32)
            sum += __expf(vv[k].y - m);
            sum += __expf(vv[k].z - m);
            sum += __expf(vv[k].w - m);
        }
        #pragma unroll
        for (int off = 16; off > 0; off >>= 1)
            sum += __shfl_xor(sum, off, 32);
        if (lane == 0) scal[s] = m + __logf(sum);
    }
    __syncthreads();

    // --- Phase B: WMMA broadcast tiles: D[M,N] = cnt[c0+N] - scal[s0+M] ---
    // A: 16x32 f16, A[M,0]=1 else 0  -> lane<16: half element 0 = 1 (M=lane,K=0)
    // B: 32x16 f16, B[0,N]=cnt[c0+N] -> lane<16: half element 0 (K=0,N=lane)
    // counts are integers <= 63: exact in f16; accumulation is f32 -> exact.
    v16h a = {};
    if (lane < 16) a[0] = (_Float16)1.0f;
    const int hi = lane >> 4;

    for (int tile = wave; tile < 256; tile += 16) {
        const int s0 = (tile >> 6) << 4;                 // s-tile * 16
        const int c0 = (tile & 63) << 4;                 // c-tile * 16
        v16h bmat = {};
        if (lane < 16) bmat[0] = (_Float16)cnt[c0 + lane];
        v8f acc;
        #pragma unroll
        for (int j = 0; j < 8; ++j)
            acc[j] = -scal[s0 + j + 8 * hi];             // C = -(max+lse) per row
        acc = __builtin_amdgcn_wmma_f32_16x16x32_f16(
            /*neg_a=*/false, a, /*neg_b=*/false, bmat,
            /*c_mod=*/(short)0, acc, /*reuse_a=*/false, /*reuse_b=*/false);

        const int cc    = c0 + (lane & 15);              // this lane's column
        const unsigned sel  = 1u << (cc & 31);
        const int      widx = cc >> 5;
        #pragma unroll
        for (int j = 0; j < 8; ++j) {
            const int s = s0 + j + 8 * hi;               // accumulator row -> s
            const unsigned w = bmask[s * 32 + widx];     // one broadcast ds load
            float v = acc[j];
            v = (w & sel) ? v - NEGV : v;                // branchless patch
            out[(((size_t)n * Ss + s) << 10) + cc] = v;
        }
    }

    // --- v output: zeros [N,S,1] appended after pi ---
    if (tid < Ss)
        out[(size_t)Nn * Ss * Cc + (size_t)n * Ss + tid] = 0.0f;
}

// ---------------------------------------------------------------------------
// Host launcher. Inputs: d_in[0]=null_w (unused by outputs), d_in[1]=ban_ids
// (int32 [N,S,B]), d_in[2]=item_ids (int32 [N,S]). Output: pi [N,S,C] f32
// followed by v [N,S,1] f32. Workspace: counts float[N*C] = 2 MB.
// ---------------------------------------------------------------------------
extern "C" void kernel_launch(void* const* d_in, const int* in_sizes, int n_in,
                              void* d_out, int out_size, void* d_ws, size_t ws_size,
                              hipStream_t stream) {
    const int* ban_ids  = (const int*)d_in[1];
    const int* item_ids = (const int*)d_in[2];
    float*     out      = (float*)d_out;
    float*     counts   = (float*)d_ws;

    counts_kernel<<<dim3(Nn), dim3(128), 0, stream>>>(item_ids, counts);
    softmax_kernel<<<dim3(Nn), dim3(512), 0, stream>>>(counts, ban_ids, out);
}
